// Decoder_11330123727470
// MI455X (gfx1250) — matrix operand order repaired and checked by reading, not compiled, not run on hardware
//
#include <hip/hip_runtime.h>
#include <hip/hip_bf16.h>

typedef __attribute__((ext_vector_type(16))) __bf16 v16bf;
typedef __attribute__((ext_vector_type(8)))  __bf16 v8bf;
typedef __attribute__((ext_vector_type(2)))  __bf16 bf16x2;
typedef __attribute__((ext_vector_type(8)))  float  v8f;
typedef __attribute__((ext_vector_type(2)))  float  f32x2;

#define Tt 128
#define Uu 1024
#define Dd 512
#define Bb 32
#define Vv 32000
#define G4 4096   // 4*U

// Build a 16-elem bf16 fragment from two contiguous 8-elem (16B) runs
// (LDS: 2x ds_load_b128; global: 2x global_load_b128).
__device__ __forceinline__ v16bf ld_frag16(const __bf16* p0, const __bf16* p1) {
  v8bf a = *(const v8bf*)p0;
  v8bf b = *(const v8bf*)p1;
  return __builtin_shufflevector(a, b, 0, 1, 2, 3, 4, 5, 6, 7,
                                 8, 9, 10, 11, 12, 13, 14, 15);
}

// ---------------------------------------------------------------------------
// One-time weight pack: W[K][N] f32  ->  WT[N][K] bf16 (WMMA-native layout).
// ---------------------------------------------------------------------------
template <int K, int N>
__global__ __launch_bounds__(256) void transpose_to_bf16(
    const float* __restrict__ W, __bf16* __restrict__ WT)
{
  __shared__ float tile[32][33];
  const int tid = threadIdx.x;
  const int n0 = blockIdx.x * 32;
  const int k0 = blockIdx.y * 32;
  #pragma unroll
  for (int i = 0; i < 4; ++i) {
    int idx = tid + i * 256;
    int r = idx >> 5, c = idx & 31;
    tile[r][c] = W[(size_t)(k0 + r) * N + n0 + c];
  }
  __syncthreads();
  #pragma unroll
  for (int i = 0; i < 2; ++i) {
    int idx = tid + i * 256;
    int n = idx >> 4, kp = idx & 15;
    f32x2 v = { tile[2 * kp][n], tile[2 * kp + 1][n] };
    *(bf16x2*)&WT[(size_t)(n0 + n) * K + k0 + 2 * kp] =
        __builtin_convertvector(v, bf16x2);
  }
}

// ---------------------------------------------------------------------------
// GEMM: C[M,N] = A[M,K](bf16) @ WT[N,K](bf16, pre-transposed) + bias[N](f32).
// BM=64, BN=128; 256 thr = 8 waves; wave = 2x2 WMMA tiles.
// A staged in LDS in 128-wide K-stages (1 barrier pair per 16 WMMAs);
// B fragments streamed straight from global (no intra-block reuse).
// ---------------------------------------------------------------------------
template <int N, int K>
__global__ __launch_bounds__(256) void gemm_bf16_wmma(
    const __bf16* __restrict__ A, const __bf16* __restrict__ WT,
    const float* __restrict__ bias, float* __restrict__ C)
{
  __shared__ __align__(16) __bf16 sA[64][136];   // [m][k-stage], pitch 272B

  const int tid  = threadIdx.x;
  const int lane = tid & 31;
  const int wave = tid >> 5;
  const int wm   = wave >> 2;          // 0..1
  const int wn   = wave & 3;           // 0..3
  const int m0   = blockIdx.y * 64;
  const int n0   = blockIdx.x * 128;
  const int row  = lane & 15;
  const int hi   = lane >> 4;          // 0 or 1

  // A staging: four 16B copies per thread per 128-wide K-stage
  const __bf16* pA  = A + (size_t)(m0 + (tid >> 2)) * K + (tid & 3) * 8;
  __bf16*       sAp = &sA[tid >> 2][(tid & 3) * 8];
  const __bf16* arp[2] = { sA[wm * 32 + row], sA[wm * 32 + 16 + row] };

  // B fragments: 32 contiguous bytes per lane per tile, direct from global
  const __bf16* pB0 = WT + (size_t)(n0 + wn * 32 + row) * K + hi * 16;
  const __bf16* pB1 = WT + (size_t)(n0 + wn * 32 + 16 + row) * K + hi * 16;

  v8f acc[2][2] = {};

  for (int ks = 0; ks < K; ks += 128) {
    #pragma unroll
    for (int i = 0; i < 4; ++i)
      *(v8bf*)(sAp + i * 32) = *(const v8bf*)(pA + i * 32);
    pA += 128;
    if (ks + 128 < K) {
      __builtin_prefetch(pB0 + 128, 0, 1);
      __builtin_prefetch(pB1 + 128, 0, 1);
    }
    __syncthreads();

    #pragma unroll
    for (int kk = 0; kk < 128; kk += 32) {
      v16bf bfrag0 = ld_frag16(pB0 + kk, pB0 + kk + 8);
      v16bf bfrag1 = ld_frag16(pB1 + kk, pB1 + kk + 8);
      v16bf afrag0 = ld_frag16(arp[0] + kk + hi * 16, arp[0] + kk + hi * 16 + 8);
      v16bf afrag1 = ld_frag16(arp[1] + kk + hi * 16, arp[1] + kk + hi * 16 + 8);

      acc[0][0] = __builtin_amdgcn_wmma_f32_16x16x32_bf16(
          false, afrag0, false, bfrag0, (short)0, acc[0][0], false, false);
      acc[0][1] = __builtin_amdgcn_wmma_f32_16x16x32_bf16(
          false, afrag0, false, bfrag1, (short)0, acc[0][1], false, false);
      acc[1][0] = __builtin_amdgcn_wmma_f32_16x16x32_bf16(
          false, afrag1, false, bfrag0, (short)0, acc[1][0], false, false);
      acc[1][1] = __builtin_amdgcn_wmma_f32_16x16x32_bf16(
          false, afrag1, false, bfrag1, (short)0, acc[1][1], false, false);
    }
    pB0 += 128;
    pB1 += 128;
    __syncthreads();
  }

  // C/D layout: VGPR r -> M = r + 8*lane[4]; N = lane[3:0]
  const int nn = n0 + wn * 32 + row;
  const float bv0 = bias[nn], bv1 = bias[nn + 16];
  float* pC = C + (size_t)(m0 + wm * 32 + hi * 8) * N + nn;
  #pragma unroll
  for (int i = 0; i < 2; ++i) {
    float* pr = pC + (size_t)(i * 16) * N;
    #pragma unroll
    for (int r = 0; r < 8; ++r) {
      pr[0]  = acc[i][0][r] + bv0;
      pr[16] = acc[i][1][r] + bv1;
      pr += N;
    }
  }
}

// ---------------------------------------------------------------------------
// Recurrent step: z[32,4096] = Xt + hbf[32,1024](bf16) @ WhT[4096,1024](bf16).
// The ENTIRE h matrix (64KB) is preloaded into LDS once (320KB/WGP), so the
// K-loop has no barriers and no LDS stores: pure WhT streaming + WMMA.
// 32 blocks x 256 thr; wave owns 16 cols, 2 WMMA tiles (M 0-15 / 16-31).
// ---------------------------------------------------------------------------
__global__ __launch_bounds__(256) void lstm_step_gemm(
    const float* __restrict__ Xt, const __bf16* __restrict__ hbf,
    const __bf16* __restrict__ WhT, float* __restrict__ z)
{
  __shared__ __align__(16) __bf16 sH[32][1032];  // full h, pitch 2064B (16B-aligned)

  const int tid  = threadIdx.x;
  const int lane = tid & 31;
  const int wave = tid >> 5;
  const int nb0  = blockIdx.x * 128;
  const int row  = lane & 15;
  const int hi   = lane >> 4;

  // preload h: 32x1024 bf16, 8 threads per row, 16 b128 copies per thread
  {
    const int m   = tid >> 3;          // 0..31
    const int seg = (tid & 7) * 128;   // contiguous 128-elem run per thread
    const __bf16* src = hbf + m * Uu + seg;
    __bf16* dst = &sH[m][seg];
    #pragma unroll
    for (int i = 0; i < 16; ++i)
      *(v8bf*)(dst + i * 8) = *(const v8bf*)(src + i * 8);
  }
  __syncthreads();

  const __bf16* arp[2] = { sH[row], sH[16 + row] };
  const __bf16* pB = WhT + (size_t)(nb0 + wave * 16 + row) * Uu + hi * 16;

  v8f acc[2] = {};

  for (int kc = 0; kc < Uu; kc += 32) {
    v16bf bfrag = ld_frag16(pB, pB + 8);
    if (kc + 32 < Uu)
      __builtin_prefetch(pB + 32, 0, 1);
    pB += 32;

    #pragma unroll
    for (int i = 0; i < 2; ++i) {
      v16bf afrag = ld_frag16(arp[i] + kc + hi * 16, arp[i] + kc + hi * 16 + 8);
      acc[i] = __builtin_amdgcn_wmma_f32_16x16x32_bf16(
          false, afrag, false, bfrag, (short)0, acc[i], false, false);
    }
  }

  const int nn = nb0 + wave * 16 + row;
  #pragma unroll
  for (int i = 0; i < 2; ++i)
    #pragma unroll
    for (int r = 0; r < 8; ++r) {
      int b = i * 16 + hi * 8 + r;       // batch row
      z[b * G4 + nn] = acc[i][r] + Xt[(size_t)b * (Tt * G4) + nn];
    }
}

// ---------------------------------------------------------------------------
// Gate math + Keras mask-carry; also produces bf16 h for the next step's GEMM
// and bf16 y for the next layer's input GEMM.
// ---------------------------------------------------------------------------
__global__ void lstm_gate(const float* __restrict__ z, const int* __restrict__ xtok,
                          float* __restrict__ h, float* __restrict__ c,
                          float* __restrict__ yprev, __bf16* __restrict__ hbf,
                          __bf16* __restrict__ yout, int t)
{
  int idx = blockIdx.x * blockDim.x + threadIdx.x;
  int b = idx >> 10, u = idx & (Uu - 1);
  const float* zb = z + b * G4;
  float ig = 1.f / (1.f + __expf(-zb[u]));
  float fg = 1.f / (1.f + __expf(-zb[Uu + u]));
  float gg = tanhf(zb[2 * Uu + u]);
  float og = 1.f / (1.f + __expf(-zb[3 * Uu + u]));
  float cn = fg * c[idx] + ig * gg;
  float hn = og * tanhf(cn);
  bool  m  = xtok[b * Tt + t] != 0;
  float h2 = m ? hn : h[idx];
  float c2 = m ? cn : c[idx];
  float y  = m ? hn : yprev[idx];
  h[idx] = h2;  c[idx] = c2;  yprev[idx] = y;
  hbf[idx] = (__bf16)h2;
  yout[(size_t)(b * Tt + t) * Uu + u] = (__bf16)y;
}

__global__ void embed_kernel(const int* __restrict__ x, const float* __restrict__ emb,
                             __bf16* __restrict__ e)
{
  size_t idx = (size_t)blockIdx.x * blockDim.x + threadIdx.x;  // B*T*D
  int row = (int)(idx >> 9);   // /D (512)
  int d   = (int)(idx & (Dd - 1));
  int tok = x[row];
  e[idx] = (__bf16)emb[(size_t)tok * Dd + d];
}

__global__ void init_state(const float* __restrict__ h0, const float* __restrict__ c0,
                           float* __restrict__ h, float* __restrict__ c,
                           float* __restrict__ yp, __bf16* __restrict__ hbf)
{
  int idx = blockIdx.x * blockDim.x + threadIdx.x;  // 0..65535 (L*B*U)
  float hv = h0[idx];
  h[idx] = hv;
  c[idx] = c0[idx];
  yp[idx] = 0.f;
  hbf[idx] = (__bf16)hv;
}

__global__ void write_states(const float* __restrict__ h, const float* __restrict__ c,
                             float* __restrict__ out)
{
  int idx = blockIdx.x * blockDim.x + threadIdx.x;  // 0..131071
  out[idx] = (idx < 2 * Bb * Uu) ? h[idx] : c[idx - 2 * Bb * Uu];
}

extern "C" void kernel_launch(void* const* d_in, const int* in_sizes, int n_in,
                              void* d_out, int out_size, void* d_ws, size_t ws_size,
                              hipStream_t stream) {
  const int*   x    = (const int*)d_in[0];
  const float* h0   = (const float*)d_in[1];
  const float* c0   = (const float*)d_in[2];
  const float* emb  = (const float*)d_in[3];
  const float* W0x  = (const float*)d_in[4];
  const float* W0h  = (const float*)d_in[5];
  const float* b0   = (const float*)d_in[6];
  const float* W1x  = (const float*)d_in[7];
  const float* W1h  = (const float*)d_in[8];
  const float* b1   = (const float*)d_in[9];
  const float* Wout = (const float*)d_in[10];
  const float* bout = (const float*)d_in[11];

  float* logits     = (float*)d_out;                       // [B*T, V]
  float* out_states = logits + (size_t)Bb * Tt * Vv;       // h_stack then c_stack

  // f32 region
  float* ws     = (float*)d_ws;
  float* Xpre   = ws;                                   // B*T*4U  = 16,777,216
  float* zbuf   = Xpre + (size_t)Bb * Tt * G4;          // B*4U    =    131,072
  float* hstate = zbuf + (size_t)Bb * G4;               // L*B*U   =     65,536
  float* cstate = hstate + 2 * Bb * Uu;                 // L*B*U   =     65,536
  float* yprev  = cstate + 2 * Bb * Uu;                 // L*B*U   =     65,536
  // bf16 region (16B-aligned: all preceding counts are multiples of 4 floats)
  __bf16* e_bf  = (__bf16*)(yprev + 2 * Bb * Uu);
  __bf16* y0bf  = e_bf + (size_t)Bb * Tt * Dd;          // B*T*U
  __bf16* y1bf  = y0bf + (size_t)Bb * Tt * Uu;          // B*T*U
  __bf16* hbf   = y1bf + (size_t)Bb * Tt * Uu;          // L*B*U
  __bf16* W0xT  = hbf + 2 * Bb * Uu;                    // [4096][512]
  __bf16* W0hT  = W0xT + (size_t)G4 * Dd;               // [4096][1024]
  __bf16* W1xT  = W0hT + (size_t)G4 * Uu;               // [4096][1024]
  __bf16* W1hT  = W1xT + (size_t)G4 * Uu;               // [4096][1024]
  __bf16* WoutT = W1hT + (size_t)G4 * Uu;               // [32000][1024]

  // one-time weight packing (transpose + bf16)
  transpose_to_bf16<Dd, G4><<<dim3(G4 / 32, Dd / 32), 256, 0, stream>>>(W0x, W0xT);
  transpose_to_bf16<Uu, G4><<<dim3(G4 / 32, Uu / 32), 256, 0, stream>>>(W0h, W0hT);
  transpose_to_bf16<Uu, G4><<<dim3(G4 / 32, Uu / 32), 256, 0, stream>>>(W1x, W1xT);
  transpose_to_bf16<Uu, G4><<<dim3(G4 / 32, Uu / 32), 256, 0, stream>>>(W1h, W1hT);
  transpose_to_bf16<Uu, Vv><<<dim3(Vv / 32, Uu / 32), 256, 0, stream>>>(Wout, WoutT);

  init_state<<<(2 * Bb * Uu) / 256, 256, 0, stream>>>(h0, c0, hstate, cstate, yprev, hbf);
  embed_kernel<<<(Bb * Tt * Dd) / 256, 256, 0, stream>>>(x, emb, e_bf);

  // X0 = e @ W0x + b0   (M=4096, N=4096, K=512) — hoisted out of the recurrence
  gemm_bf16_wmma<G4, Dd><<<dim3(G4 / 128, (Bb * Tt) / 64), 256, 0, stream>>>(
      e_bf, W0xT, b0, Xpre);

  for (int t = 0; t < Tt; ++t) {
    lstm_step_gemm<<<G4 / 128, 256, 0, stream>>>(Xpre + (size_t)t * G4, hbf, W0hT, zbuf);
    lstm_gate<<<(Bb * Uu) / 256, 256, 0, stream>>>(zbuf, x, hstate, cstate, yprev,
                                                   hbf, y0bf, t);
  }

  // X1 = y0 @ W1x + b1   (reuse Xpre buffer; X0 dead after layer-0 loop)
  gemm_bf16_wmma<G4, Uu><<<dim3(G4 / 128, (Bb * Tt) / 64), 256, 0, stream>>>(
      y0bf, W1xT, b1, Xpre);

  for (int t = 0; t < Tt; ++t) {
    lstm_step_gemm<<<G4 / 128, 256, 0, stream>>>(Xpre + (size_t)t * G4,
                                                 hbf + Bb * Uu, W1hT, zbuf);
    lstm_gate<<<(Bb * Uu) / 256, 256, 0, stream>>>(zbuf, x, hstate + Bb * Uu,
                                                   cstate + Bb * Uu, yprev + Bb * Uu,
                                                   hbf + Bb * Uu, y1bf, t);
  }

  // logits = y1 @ Wout + bout   (M=4096, N=32000, K=1024) — the 268 GF kernel
  gemm_bf16_wmma<Vv, Uu><<<dim3(Vv / 128, (Bb * Tt) / 64), 256, 0, stream>>>(
      y1bf, WoutT, bout, logits);

  write_states<<<(4 * Bb * Uu) / 256, 256, 0, stream>>>(hstate, cstate, out_states);
}
